// LinearShift_31379031064616
// MI455X (gfx1250) — matrix operand order repaired and checked by reading, not compiled, not run on hardware
//
#include <hip/hip_runtime.h>

// LinearShift on MI455X (gfx1250):
//   out = fixed_quant(x) @ (±2^round(shift)).T + fixed_quant(bias)
// Roofline: 275 GFLOP vs ~400MB mandatory traffic @23.3TB/s (17us floor) ->
// compute-bound on the matrix pipe. f16 is exact for this data (quantized x
// has <=11 significand bits for |x|<8; weights are powers of two in
// [2^-10,2^-1]), so one V_WMMA_F32_16X16X32_F16 GEMM with f32 accumulation
// matches the f32 reference. A (64MB f16) + W (32MB f16) both fit in the
// 192MB L2, so repeated tile reads are L2 hits.

typedef _Float16 v16h __attribute__((ext_vector_type(16)));
typedef _Float16 v8h  __attribute__((ext_vector_type(8)));
typedef _Float16 v4h  __attribute__((ext_vector_type(4)));
typedef float    v8f  __attribute__((ext_vector_type(8)));

#define DELTA_INV 256.0f
#define DELTA     0.00390625f   // 2^-8

// CDNA5 async global->LDS copies (ASYNCcnt-tracked), if the toolchain
// exposes them; otherwise fall back to VGPR staging.
#if defined(__has_builtin)
#  if __has_builtin(__builtin_amdgcn_global_load_async_to_lds_b128)
#    define HAVE_ASYNC_LDS 1
#  endif
#endif
#ifndef HAVE_ASYNC_LDS
#  define HAVE_ASYNC_LDS 0
#endif

// Payload type per the builtin's signature: 128-bit int vector.
typedef int b128v __attribute__((vector_size(16)));
typedef __attribute__((address_space(1))) b128v g_b128;   // global (AS1)
typedef __attribute__((address_space(3))) b128v l_b128;   // LDS (AS3)

__device__ __forceinline__ void wait_asynccnt0() {
#if defined(__has_builtin) && __has_builtin(__builtin_amdgcn_s_wait_asynccnt)
  __builtin_amdgcn_s_wait_asynccnt(0);
#else
  asm volatile("s_wait_asynccnt 0x0" ::: "memory");
#endif
}

__device__ __forceinline__ float quant_fixed(float x) {
  float r = floorf(__builtin_fmaf(x, DELTA_INV, 0.5f));
  r = fminf(fmaxf(r, -32768.0f), 32767.0f);
  return r * DELTA;
}

// ---------------- Pass 1: quantize activations to f16 ----------------
__global__ void quant_x_f16(const float4* __restrict__ x, v4h* __restrict__ xq, int n4) {
  int i = blockIdx.x * blockDim.x + threadIdx.x;
  if (i >= n4) return;
  float4 v = x[i];
  v4h o;
  o[0] = (_Float16)quant_fixed(v.x);
  o[1] = (_Float16)quant_fixed(v.y);
  o[2] = (_Float16)quant_fixed(v.z);
  o[3] = (_Float16)quant_fixed(v.w);
  xq[i] = o;
}

// ------------- Pass 2: materialize weights v = ±2^round(shift) -------------
// round(sign) in {-1,-0,0}; negative iff it rounds to -1 (matches JAX %2 rule).
__device__ __forceinline__ _Float16 wval(float sh, float sg) {
  float w = exp2f(rintf(sh));
  return (_Float16)((rintf(sg) != 0.0f) ? -w : w);
}

__global__ void make_w_f16(const float4* __restrict__ shift,
                           const float4* __restrict__ sign,
                           v4h* __restrict__ wq, int n4) {
  int i = blockIdx.x * blockDim.x + threadIdx.x;
  if (i >= n4) return;
  float4 sh = shift[i];
  float4 sg = sign[i];
  v4h o;
  o[0] = wval(sh.x, sg.x);
  o[1] = wval(sh.y, sg.y);
  o[2] = wval(sh.z, sg.z);
  o[3] = wval(sh.w, sg.w);
  wq[i] = o;
}

// ---------------- Pass 3: WMMA GEMM ----------------
// A: M x K f16 (row-major), W: N x K f16 (row-major) -> out[m][n] = sum_k A*W.
// Block tile 128x128, TK=32. 256 threads = 8 waves, each wave owns 32(M)x64(N)
// = 2x4 grid of 16x16 WMMA tiles. Double-buffered LDS; async DMA (or staged)
// global->LDS; one barrier per K-step.
#define TM 128
#define TN 128
#define TK 32
#define KST 40   // LDS row stride in halfs (32 + 8 pad, keeps 16B alignment)

__global__ __launch_bounds__(256) void gemm_wmma(
    const _Float16* __restrict__ Aq,   // M x K
    const _Float16* __restrict__ Wq,   // N x K
    const float*    __restrict__ bias, // N
    float*          __restrict__ out,  // M x N
    int M, int N, int K)
{
  __shared__ __align__(16) _Float16 sh[2][2][TM * KST];  // [buf][A/B]

  const int tid  = threadIdx.x;
  const int lane = tid & 31;
  const int wv   = tid >> 5;
  const int nbx  = N / TN;
  const int bx   = blockIdx.x % nbx;
  const int by   = blockIdx.x / nbx;
  const int row0 = by * TM;
  const int col0 = bx * TN;
  const int waveM = (wv >> 1) * 32;   // 4 waves along M
  const int waveN = (wv & 1) * 64;    // 2 waves along N

  // Global->LDS tiling: per K-step each matrix tile is 128 rows x 64B
  // = 512 x 16B chunks; 2 chunks per thread per matrix.
  const int r0 = tid >> 2;            // rows 0..63
  const int p0 = (tid & 3) * 8;       // half offset within row: 0,8,16,24
  const int r1 = r0 + 64;             // rows 64..127

  const _Float16* gA0 = Aq + (size_t)(row0 + r0) * K + p0;
  const _Float16* gA1 = Aq + (size_t)(row0 + r1) * K + p0;
  const _Float16* gB0 = Wq + (size_t)(col0 + r0) * K + p0;
  const _Float16* gB1 = Wq + (size_t)(col0 + r1) * K + p0;

  // Fragment addressing (ISA 7.12.2, 16-bit matrices, wave32):
  //  A 16x32: lanes 0-15 -> row M=lane, K {0..7,16..23}; lanes 16-31 -> {8..15,24..31}
  //  B 32x16: lanes 0-15 -> col N=lane, K {0..7,16..23}; lanes 16-31 -> {8..15,24..31}
  const int fr  = lane & 15;
  const int koA = (lane < 16) ? 0 : 8;
  const int koB = (lane < 16) ? 0 : 8;

  v8f acc[2][4];
  #pragma unroll
  for (int mi = 0; mi < 2; ++mi)
    #pragma unroll
    for (int ni = 0; ni < 4; ++ni)
      #pragma unroll
      for (int r = 0; r < 8; ++r) acc[mi][ni][r] = 0.0f;

  const int S = K / TK;  // 128 K-steps

#if HAVE_ASYNC_LDS
  // ----- async global->LDS double buffering (GLOBAL_LOAD_ASYNC_TO_LDS_B128) -----
  auto issue_tile = [&](int ksIdx, int buf) {
    const int kk = ksIdx * TK;
    __builtin_amdgcn_global_load_async_to_lds_b128(
        (g_b128*)(gA0 + kk), (l_b128*)&sh[buf][0][r0 * KST + p0], 0, 0);
    __builtin_amdgcn_global_load_async_to_lds_b128(
        (g_b128*)(gA1 + kk), (l_b128*)&sh[buf][0][r1 * KST + p0], 0, 0);
    __builtin_amdgcn_global_load_async_to_lds_b128(
        (g_b128*)(gB0 + kk), (l_b128*)&sh[buf][1][r0 * KST + p0], 0, 0);
    __builtin_amdgcn_global_load_async_to_lds_b128(
        (g_b128*)(gB1 + kk), (l_b128*)&sh[buf][1][r1 * KST + p0], 0, 0);
  };

  issue_tile(0, 0);
  wait_asynccnt0();
  __syncthreads();
#else
  uint4 ra0 = *(const uint4*)gA0;
  uint4 ra1 = *(const uint4*)gA1;
  uint4 rb0 = *(const uint4*)gB0;
  uint4 rb1 = *(const uint4*)gB1;
#endif

  #pragma unroll 1
  for (int ks = 0; ks < S; ++ks) {
    const int buf = ks & 1;

#if HAVE_ASYNC_LDS
    if (ks + 1 < S) issue_tile(ks + 1, buf ^ 1);  // DMA next tile during compute
#else
    *(uint4*)&sh[buf][0][r0 * KST + p0] = ra0;
    *(uint4*)&sh[buf][0][r1 * KST + p0] = ra1;
    *(uint4*)&sh[buf][1][r0 * KST + p0] = rb0;
    *(uint4*)&sh[buf][1][r1 * KST + p0] = rb1;
    __syncthreads();

    if (ks + 1 < S) {
      const int k1 = (ks + 1) * TK;
      ra0 = *(const uint4*)(gA0 + k1);
      ra1 = *(const uint4*)(gA1 + k1);
      rb0 = *(const uint4*)(gB0 + k1);
      rb1 = *(const uint4*)(gB1 + k1);
      if (ks + 2 < S) {  // global_prefetch_b8 two steps ahead
        __builtin_prefetch(gA0 + (ks + 2) * TK, 0, 1);
        __builtin_prefetch(gB0 + (ks + 2) * TK, 0, 1);
      }
    }
#endif

    const _Float16* sA = &sh[buf][0][0];
    const _Float16* sB = &sh[buf][1][0];

    v16h af[2], bf[4];
    #pragma unroll
    for (int mi = 0; mi < 2; ++mi) {
      const _Float16* pr = sA + (waveM + mi * 16 + fr) * KST;
      v8h lo = *(const v8h*)(pr + koA);
      v8h hi = *(const v8h*)(pr + 16 + koA);
      af[mi] = __builtin_shufflevector(lo, hi, 0,1,2,3,4,5,6,7,8,9,10,11,12,13,14,15);
    }
    #pragma unroll
    for (int ni = 0; ni < 4; ++ni) {
      const _Float16* pr = sB + (waveN + ni * 16 + fr) * KST + koB;
      v8h lo = *(const v8h*)(pr);
      v8h hi = *(const v8h*)(pr + 16);
      bf[ni] = __builtin_shufflevector(lo, hi, 0,1,2,3,4,5,6,7,8,9,10,11,12,13,14,15);
    }

    #pragma unroll
    for (int mi = 0; mi < 2; ++mi)
      #pragma unroll
      for (int ni = 0; ni < 4; ++ni)
        acc[mi][ni] = __builtin_amdgcn_wmma_f32_16x16x32_f16(
            false, af[mi], false, bf[ni], (short)0, acc[mi][ni], false, false);

#if HAVE_ASYNC_LDS
    if (ks + 1 < S) {
      wait_asynccnt0();   // next buffer fully in LDS (this wave's DMAs)
      __syncthreads();    // ...and everyone else's; also fences buf reuse
    }
#endif
  }

  // Epilogue: C/D layout (VGPR r: lanes 0-15 -> M=r, lanes 16-31 -> M=8+r;
  // N = lane%16). Add quantized bias, store f32.
  const int colL = lane & 15;
  const int mOff = (lane < 16) ? 0 : 8;
  #pragma unroll
  for (int ni = 0; ni < 4; ++ni) {
    const int gn = col0 + waveN + ni * 16 + colL;
    const float bq = quant_fixed(bias[gn]);
    #pragma unroll
    for (int mi = 0; mi < 2; ++mi) {
      #pragma unroll
      for (int r = 0; r < 8; ++r) {
        const int gm = row0 + waveM + mi * 16 + mOff + r;
        out[(size_t)gm * N + gn] = acc[mi][ni][r] + bq;
      }
    }
  }
}

extern "C" void kernel_launch(void* const* d_in, const int* in_sizes, int n_in,
                              void* d_out, int out_size, void* d_ws, size_t ws_size,
                              hipStream_t stream) {
  const float* x     = (const float*)d_in[0];
  const float* shift = (const float*)d_in[1];
  const float* sign  = (const float*)d_in[2];
  const float* bias  = (const float*)d_in[3];
  float* out = (float*)d_out;

  const int K = 4096;
  const int N = in_sizes[3];          // 4096 (bias length = OUT)
  const int M = in_sizes[0] / K;      // 8192

  _Float16* xq = (_Float16*)d_ws;                 // M*K f16 = 64 MB
  _Float16* wq = xq + (size_t)M * K;              // N*K f16 = 32 MB

  {
    int n4 = (M * K) / 4;
    quant_x_f16<<<n4 / 256, 256, 0, stream>>>((const float4*)x, (v4h*)xq, n4);
  }
  {
    int n4 = (N * K) / 4;
    make_w_f16<<<n4 / 256, 256, 0, stream>>>((const float4*)shift,
                                             (const float4*)sign, (v4h*)wq, n4);
  }
  {
    dim3 grid((N / TN) * (M / TM));   // 32 * 64 = 2048 workgroups
    gemm_wmma<<<grid, 256, 0, stream>>>(xq, wq, bias, out, M, N, K);
  }
}
